// RSSM_15023795601717
// MI455X (gfx1250) — matrix operand order repaired and checked by reading, not compiled, not run on hardware
//
#include <hip/hip_runtime.h>
#include <hip/hip_bf16.h>

// ---------------------------------------------------------------------------
// RSSM GRU step for MI455X (gfx1250, wave32, WMMA).
// - All heavy GEMMs: v_wmma_f32_16x16x32_bf16, f32 accumulate.
// - f32 weights converted to bf16 during global->LDS staging.
// - Double-buffered LDS: one barrier per k-step, stage t+1 overlaps WMMA t.
// - bf16 A tiles staged via async global->LDS (ASYNCcnt) when the builtin
//   exists; falls back to load+ds_store otherwise.
// - Final Wb2 GEMM fused with the sigmoid/tanh GRU gate (no (B,3D) buffer).
// ---------------------------------------------------------------------------

#ifndef __has_builtin
#define __has_builtin(x) 0
#endif

#if __has_builtin(__builtin_amdgcn_global_load_async_to_lds_b128)
#define USE_ASYNC_LDS 1
#else
#define USE_ASYNC_LDS 0
#endif

typedef __bf16 bf16_t;
typedef __attribute__((ext_vector_type(16))) __bf16        v16bf;
typedef __attribute__((ext_vector_type(8)))  float          v8f;
typedef __attribute__((ext_vector_type(8)))  unsigned short ushort8;
typedef __attribute__((ext_vector_type(4)))  unsigned short us4;
typedef __attribute__((ext_vector_type(4)))  float          fx4;
typedef __attribute__((ext_vector_type(4)))  int            i4x;   // async-LDS builtin arg type

#define BATCH 1024   // 16*64
#define DDIM  8192
#define HDIM  1024
#define SDIM  2048   // 32*64
#define GRP   8

#define BM 128
#define BN 64
#define BK 32
#define LDSK (BK + 8)   // 40 bf16 = 80 B row stride (16B aligned, conflict-free)

static __device__ __forceinline__ v8f wmma_bf16(const v16bf& a, const v16bf& b, v8f c) {
  return __builtin_amdgcn_wmma_f32_16x16x32_bf16(false, a, false, b, (short)0, c, false, false);
}
static __device__ __forceinline__ float sigmoidf_(float x) {
  return 1.0f / (1.0f + __expf(-x));
}

// Copy 16 bytes of bf16 A-tile data global -> LDS.
static __device__ __forceinline__ void stage_a16(const bf16_t* gp, bf16_t* lp) {
#if USE_ASYNC_LDS
  __builtin_amdgcn_global_load_async_to_lds_b128((i4x*)gp, (i4x*)lp, 0, 0);
#else
  *(ushort8*)lp = *(const ushort8*)gp;
#endif
}

// Wait for this wave's staging traffic, then workgroup barrier.
static __device__ __forceinline__ void stage_sync() {
#if USE_ASYNC_LDS
#if __has_builtin(__builtin_amdgcn_s_wait_asynccnt)
  __builtin_amdgcn_s_wait_asynccnt(0);
#else
  asm volatile("s_wait_asynccnt 0x0" ::: "memory");
#endif
#endif
  __syncthreads();
}

// ---------------------------------------------------------------------------
// f32 -> bf16 bulk convert (n multiple of 4)
// ---------------------------------------------------------------------------
__global__ __launch_bounds__(256) void f32_to_bf16_kernel(
    const float* __restrict__ in, bf16_t* __restrict__ out, int n) {
  int i = (blockIdx.x * 256 + threadIdx.x) * 4;
  if (i >= n) return;
  fx4 v = *(const fx4*)(in + i);
  bf16_t t[4];
#pragma unroll
  for (int j = 0; j < 4; ++j) t[j] = (bf16_t)v[j];
  *(us4*)(out + i) = *(const us4*)t;
}

// ---------------------------------------------------------------------------
// Action branch: K=2 "GEMM" as elementwise; writes pre3H[:, 2048:3072] (f32)
// ---------------------------------------------------------------------------
__global__ __launch_bounds__(256) void action_branch_kernel(
    const float* __restrict__ act, const float* __restrict__ W3,
    const float* __restrict__ b3, float* __restrict__ pre /* B x 3072 */) {
  int idx = blockIdx.x * 256 + threadIdx.x;    // BATCH*HDIM threads
  int b = idx >> 10, h = idx & 1023;
  float a0 = act[b * 2 + 0], a1 = act[b * 2 + 1];
  a0 = a0 / fmaxf(1.0f, fabsf(a0));
  a1 = a1 / fmaxf(1.0f, fabsf(a1));
  pre[(size_t)b * 3072 + 2048 + h] = a0 * W3[h * 2 + 0] + a1 * W3[h * 2 + 1] + b3[h];
}

// ---------------------------------------------------------------------------
// Row-wise RMSNorm (+scale) + SiLU, f32 in -> bf16 out. One block per row.
// ---------------------------------------------------------------------------
__global__ __launch_bounds__(256) void rmsnorm_silu_kernel(
    const float* __restrict__ in, const float* __restrict__ scale,
    bf16_t* __restrict__ out, int cols, int ldin, int ldout) {
  __shared__ float red[256];
  const int row = blockIdx.x, tid = threadIdx.x;
  const float* rp = in + (size_t)row * ldin;
  float s = 0.f;
  for (int c = tid; c < cols; c += 256) { float x = rp[c]; s += x * x; }
  red[tid] = s;
  __syncthreads();
  for (int off = 128; off > 0; off >>= 1) {
    if (tid < off) red[tid] += red[tid + off];
    __syncthreads();
  }
  const float rms = rsqrtf(red[0] / (float)cols + 1e-6f);
  bf16_t* op = out + (size_t)row * ldout;
  for (int c = tid; c < cols; c += 256) {
    float y = rp[c] * rms * scale[c];
    op[c] = (bf16_t)(y * sigmoidf_(y));     // SiLU
  }
}

// ---------------------------------------------------------------------------
// Generic NT GEMM:  C[m,n] = sum_k A[m,k]*Bw[n,k] + bias[n]
// A bf16 (lda = K-stride), Bw f32 K-contiguous, C f32 (ldc).
// Tile: BM=128 x BN=64 x BK=32, 256 threads (8 waves, 4x2 of 32x32).
// Double-buffered LDS, one barrier per k-step.
// ---------------------------------------------------------------------------
__global__ __launch_bounds__(256) void gemm_bf16_nt_kernel(
    const bf16_t* __restrict__ A, int lda,
    const float* __restrict__ Bw, int ldb,
    const float* __restrict__ bias,
    float* __restrict__ C, int ldc, int K) {
  __shared__ __align__(16) bf16_t As[2][BM][LDSK];
  __shared__ __align__(16) bf16_t Bs[2][BN][LDSK];
  const int tid = threadIdx.x;
  const int lane = tid & 31, wave = tid >> 5;
  const int wm = wave >> 1, wn = wave & 1;          // 4 x 2 waves
  const int lm = lane & 15, lh = lane >> 4;
  const int m0 = blockIdx.y * BM, n0 = blockIdx.x * BN;

  v8f acc[2][2] = {};

  auto stage = [&](int buf, int k0) {
    // ---- A tile (bf16, K-contiguous): 128x32, 16B per chunk
#pragma unroll
    for (int i = 0; i < 2; ++i) {
      int c = tid + i * 256;                 // 512 chunks of 8 bf16
      int row = c >> 2, kc = (c & 3) << 3;
      stage_a16(A + (size_t)(m0 + row) * lda + k0 + kc, &As[buf][row][kc]);
    }
    // ---- B tile (f32 K-contiguous -> bf16): 64 n-rows x 32 k
    int n = tid >> 2, kc = (tid & 3) << 3;
    const float* p = Bw + (size_t)(n0 + n) * ldb + k0 + kc;
    fx4 x0 = *(const fx4*)p;
    fx4 x1 = *(const fx4*)(p + 4);
    bf16_t t[8];
#pragma unroll
    for (int j = 0; j < 4; ++j) { t[j] = (bf16_t)x0[j]; t[4 + j] = (bf16_t)x1[j]; }
    *(ushort8*)&Bs[buf][n][kc] = *(const ushort8*)t;
    if (k0 + 2 * BK < K) __builtin_prefetch(p + 2 * BK, 0, 1);   // global_prefetch_b8
  };

  stage(0, 0);
  stage_sync();

  const int iters = K / BK;
  for (int it = 0; it < iters; ++it) {
    const int buf = it & 1;
    if (it + 1 < iters) stage(buf ^ 1, (it + 1) * BK);

    v16bf af[2], bf[2];
#pragma unroll
    for (int mt = 0; mt < 2; ++mt) {
      const bf16_t* pa = &As[buf][wm * 32 + mt * 16 + lm][0];
      ((ushort8*)&af[mt])[0] = *(const ushort8*)(pa + 16 * lh);
      ((ushort8*)&af[mt])[1] = *(const ushort8*)(pa + 16 * lh + 8);
    }
#pragma unroll
    for (int nt = 0; nt < 2; ++nt) {
      const bf16_t* pb = &Bs[buf][wn * 32 + nt * 16 + lm][0];
      ((ushort8*)&bf[nt])[0] = *(const ushort8*)(pb + 16 * lh);
      ((ushort8*)&bf[nt])[1] = *(const ushort8*)(pb + 16 * lh + 8);
    }
#pragma unroll
    for (int mt = 0; mt < 2; ++mt)
#pragma unroll
      for (int nt = 0; nt < 2; ++nt)
        acc[mt][nt] = wmma_bf16(af[mt], bf[nt], acc[mt][nt]);
    stage_sync();
  }

#pragma unroll
  for (int mt = 0; mt < 2; ++mt)
#pragma unroll
    for (int nt = 0; nt < 2; ++nt) {
      int n = n0 + wn * 32 + nt * 16 + lm;
      float bv = bias[n];
#pragma unroll
      for (int v = 0; v < 8; ++v) {
        int m = m0 + wm * 32 + mt * 16 + v + 8 * lh;
        C[(size_t)m * ldc + n] = acc[mt][nt][v] + bv;
      }
    }
}

// ---------------------------------------------------------------------------
// Grouped GEMM 1:  h[b, g*1024+n] = sum_i X[b,g,i]*Wb1[g,i,n] + bb1[g*1024+n]
// X[b,g,i] = xcat[b,i] for i<3072 else dbf[b, g*1024 + (i-3072)].
// Wb1 is N-contiguous -> transpose during bf16 LDS store.
// ---------------------------------------------------------------------------
__global__ __launch_bounds__(256) void gemm_grouped_wb1_kernel(
    const bf16_t* __restrict__ xcat,   // B x 3072
    const bf16_t* __restrict__ dbf,    // B x 8192
    const float* __restrict__ Wb1,     // G x 4096 x 1024
    const float* __restrict__ bb1,     // 8192
    float* __restrict__ Hpre) {        // B x 8192
  __shared__ __align__(16) bf16_t As[2][BM][LDSK];
  __shared__ __align__(16) bf16_t Bs[2][BN][LDSK];
  const int tid = threadIdx.x;
  const int lane = tid & 31, wave = tid >> 5;
  const int wm = wave >> 1, wn = wave & 1;
  const int lm = lane & 15, lh = lane >> 4;
  const int g = blockIdx.z;
  const int m0 = blockIdx.y * BM, n0 = blockIdx.x * BN;   // n in [0,1024)
  const float* Bg = Wb1 + (size_t)g * 4096 * 1024;

  v8f acc[2][2] = {};

  auto stage = [&](int buf, int k0) {
    // A source switch (3072 % BK == 0 so a tile never straddles)
    const bf16_t* Abase;
    int ldaA, kk;
    if (k0 < 3072) { Abase = xcat;           ldaA = 3072; kk = k0; }
    else           { Abase = dbf + g * 1024; ldaA = 8192; kk = k0 - 3072; }
#pragma unroll
    for (int i = 0; i < 2; ++i) {
      int c = tid + i * 256;
      int row = c >> 2, kc = (c & 3) << 3;
      stage_a16(Abase + (size_t)(m0 + row) * ldaA + kk + kc, &As[buf][row][kc]);
    }
    // B tile [k][n] (N-contiguous) -> store transposed as Bs[n][k]
    int k = tid >> 3, ns = (tid & 7) << 3;
    const float* p = Bg + (size_t)(k0 + k) * 1024 + n0 + ns;
    fx4 x0 = *(const fx4*)p;
    fx4 x1 = *(const fx4*)(p + 4);
#pragma unroll
    for (int j = 0; j < 4; ++j) {
      Bs[buf][ns + j][k]     = (bf16_t)x0[j];
      Bs[buf][ns + 4 + j][k] = (bf16_t)x1[j];
    }
    if (k0 + 2 * BK < 4096) __builtin_prefetch(p + (size_t)(2 * BK) * 1024, 0, 1);
  };

  stage(0, 0);
  stage_sync();

  const int iters = 4096 / BK;
  for (int it = 0; it < iters; ++it) {
    const int buf = it & 1;
    if (it + 1 < iters) stage(buf ^ 1, (it + 1) * BK);

    v16bf af[2], bf[2];
#pragma unroll
    for (int mt = 0; mt < 2; ++mt) {
      const bf16_t* pa = &As[buf][wm * 32 + mt * 16 + lm][0];
      ((ushort8*)&af[mt])[0] = *(const ushort8*)(pa + 16 * lh);
      ((ushort8*)&af[mt])[1] = *(const ushort8*)(pa + 16 * lh + 8);
    }
#pragma unroll
    for (int nt = 0; nt < 2; ++nt) {
      const bf16_t* pb = &Bs[buf][wn * 32 + nt * 16 + lm][0];
      ((ushort8*)&bf[nt])[0] = *(const ushort8*)(pb + 16 * lh);
      ((ushort8*)&bf[nt])[1] = *(const ushort8*)(pb + 16 * lh + 8);
    }
#pragma unroll
    for (int mt = 0; mt < 2; ++mt)
#pragma unroll
      for (int nt = 0; nt < 2; ++nt)
        acc[mt][nt] = wmma_bf16(af[mt], bf[nt], acc[mt][nt]);
    stage_sync();
  }

#pragma unroll
  for (int mt = 0; mt < 2; ++mt)
#pragma unroll
    for (int nt = 0; nt < 2; ++nt) {
      int n = n0 + wn * 32 + nt * 16 + lm;
      float bv = bb1[g * 1024 + n];
#pragma unroll
      for (int v = 0; v < 8; ++v) {
        int m = m0 + wm * 32 + mt * 16 + v + 8 * lh;
        Hpre[(size_t)m * DDIM + g * 1024 + n] = acc[mt][nt][v] + bv;
      }
    }
}

// ---------------------------------------------------------------------------
// Fused Wb2 GEMM + GRU gate. For group g and output column j in [0,1024):
//   r = h_g . Wb2[g,:,j]      + bb2[g*3072 + j]
//   c = h_g . Wb2[g,:,1024+j] + bb2[g*3072 + 1024 + j]
//   u = h_g . Wb2[g,:,2048+j] + bb2[g*3072 + 2048 + j]
//   out = sig(u-1)*tanh(sig(r)*c) + (1-sig(u-1))*d
// Avoids materializing the (B,3D) f32 tensor (~200 MB of HBM traffic saved).
// ---------------------------------------------------------------------------
__global__ __launch_bounds__(256) void gemm_gate_wb2_kernel(
    const bf16_t* __restrict__ hb,     // B x 8192 (bf16)
    const float* __restrict__ Wb2,     // G x 1024 x 3072
    const float* __restrict__ bb2,     // 24576
    const float* __restrict__ deter,   // B x 8192 (f32, original input)
    float* __restrict__ out) {         // B x 8192
  __shared__ __align__(16) bf16_t As[2][BM][LDSK];
  __shared__ __align__(16) bf16_t Bs[2][3][BN][LDSK];
  const int tid = threadIdx.x;
  const int lane = tid & 31, wave = tid >> 5;
  const int wm = wave >> 1, wn = wave & 1;
  const int lm = lane & 15, lh = lane >> 4;
  const int g = blockIdx.z;
  const int m0 = blockIdx.y * BM, n0 = blockIdx.x * BN;   // j in [0,1024)
  const float* Bg = Wb2 + (size_t)g * 1024 * 3072;
  const bf16_t* Abase = hb + g * 1024;                    // lda = 8192, K = 1024

  v8f accR[2][2] = {}, accC[2][2] = {}, accU[2][2] = {};

  auto stage = [&](int buf, int k0) {
#pragma unroll
    for (int i = 0; i < 2; ++i) {
      int c = tid + i * 256;
      int row = c >> 2, kc = (c & 3) << 3;
      stage_a16(Abase + (size_t)(m0 + row) * DDIM + k0 + kc, &As[buf][row][kc]);
    }
    int k = tid >> 3, ns = (tid & 7) << 3;
#pragma unroll
    for (int s = 0; s < 3; ++s) {
      const float* p = Bg + (size_t)(k0 + k) * 3072 + s * 1024 + n0 + ns;
      fx4 x0 = *(const fx4*)p;
      fx4 x1 = *(const fx4*)(p + 4);
#pragma unroll
      for (int j = 0; j < 4; ++j) {
        Bs[buf][s][ns + j][k]     = (bf16_t)x0[j];
        Bs[buf][s][ns + 4 + j][k] = (bf16_t)x1[j];
      }
    }
  };

  stage(0, 0);
  stage_sync();

  const int iters = 1024 / BK;
  for (int it = 0; it < iters; ++it) {
    const int buf = it & 1;
    if (it + 1 < iters) stage(buf ^ 1, (it + 1) * BK);

    v16bf af[2], bfR[2], bfC[2], bfU[2];
#pragma unroll
    for (int mt = 0; mt < 2; ++mt) {
      const bf16_t* pa = &As[buf][wm * 32 + mt * 16 + lm][0];
      ((ushort8*)&af[mt])[0] = *(const ushort8*)(pa + 16 * lh);
      ((ushort8*)&af[mt])[1] = *(const ushort8*)(pa + 16 * lh + 8);
    }
#pragma unroll
    for (int nt = 0; nt < 2; ++nt) {
      const int col = wn * 32 + nt * 16 + lm;
      const bf16_t* pR = &Bs[buf][0][col][0];
      const bf16_t* pC = &Bs[buf][1][col][0];
      const bf16_t* pU = &Bs[buf][2][col][0];
      ((ushort8*)&bfR[nt])[0] = *(const ushort8*)(pR + 16 * lh);
      ((ushort8*)&bfR[nt])[1] = *(const ushort8*)(pR + 16 * lh + 8);
      ((ushort8*)&bfC[nt])[0] = *(const ushort8*)(pC + 16 * lh);
      ((ushort8*)&bfC[nt])[1] = *(const ushort8*)(pC + 16 * lh + 8);
      ((ushort8*)&bfU[nt])[0] = *(const ushort8*)(pU + 16 * lh);
      ((ushort8*)&bfU[nt])[1] = *(const ushort8*)(pU + 16 * lh + 8);
    }
#pragma unroll
    for (int mt = 0; mt < 2; ++mt)
#pragma unroll
      for (int nt = 0; nt < 2; ++nt) {
        accR[mt][nt] = wmma_bf16(af[mt], bfR[nt], accR[mt][nt]);
        accC[mt][nt] = wmma_bf16(af[mt], bfC[nt], accC[mt][nt]);
        accU[mt][nt] = wmma_bf16(af[mt], bfU[nt], accU[mt][nt]);
      }
    stage_sync();
  }

  // ---- fused GRU gate epilogue
#pragma unroll
  for (int mt = 0; mt < 2; ++mt)
#pragma unroll
    for (int nt = 0; nt < 2; ++nt) {
      int j = n0 + wn * 32 + nt * 16 + lm;            // column in [0,1024)
      float br = bb2[g * 3072 + j];
      float bc = bb2[g * 3072 + 1024 + j];
      float bu = bb2[g * 3072 + 2048 + j];
#pragma unroll
      for (int v = 0; v < 8; ++v) {
        int m = m0 + wm * 32 + mt * 16 + v + 8 * lh;
        size_t oidx = (size_t)m * DDIM + g * 1024 + j;
        float dv = deter[oidx];
        float r = accR[mt][nt][v] + br;
        float c = accC[mt][nt][v] + bc;
        float u = accU[mt][nt][v] + bu;
        float reset  = sigmoidf_(r);
        float cand   = tanhf(reset * c);
        float update = sigmoidf_(u - 1.0f);
        out[oidx] = update * cand + (1.0f - update) * dv;
      }
    }
}

// ---------------------------------------------------------------------------
// Workspace layout (bytes), total ~90.2 MB
// ---------------------------------------------------------------------------
static constexpr size_t SZ_DB   = (size_t)BATCH * DDIM * 2;   // deter bf16
static constexpr size_t SZ_SB   = (size_t)BATCH * SDIM * 2;   // stoch bf16
static constexpr size_t SZ_PRE  = (size_t)BATCH * 3072 * 4;   // pre-norm x1|x2|x3 f32
static constexpr size_t SZ_XCAT = (size_t)BATCH * 3072 * 2;   // silu(rms) bf16
static constexpr size_t SZ_HPRE = (size_t)BATCH * DDIM * 4;   // h pre-norm f32
static constexpr size_t OFF_DB   = 0;
static constexpr size_t OFF_SB   = OFF_DB + SZ_DB;
static constexpr size_t OFF_PRE  = OFF_SB + SZ_SB;
static constexpr size_t OFF_XCAT = OFF_PRE + SZ_PRE;
static constexpr size_t OFF_HPRE = OFF_XCAT + SZ_XCAT;
static constexpr size_t OFF_HB   = OFF_HPRE + SZ_HPRE;

extern "C" void kernel_launch(void* const* d_in, const int* in_sizes, int n_in,
                              void* d_out, int out_size, void* d_ws, size_t ws_size,
                              hipStream_t stream) {
  const float* deter  = (const float*)d_in[0];
  const float* stoch  = (const float*)d_in[1];
  const float* action = (const float*)d_in[2];
  const float* W1  = (const float*)d_in[3];
  const float* b1  = (const float*)d_in[4];
  const float* s1  = (const float*)d_in[5];
  const float* W2  = (const float*)d_in[6];
  const float* b2  = (const float*)d_in[7];
  const float* s2  = (const float*)d_in[8];
  const float* W3  = (const float*)d_in[9];
  const float* b3  = (const float*)d_in[10];
  const float* s3  = (const float*)d_in[11];
  const float* Wb1 = (const float*)d_in[12];
  const float* bb1 = (const float*)d_in[13];
  const float* sb1 = (const float*)d_in[14];
  const float* Wb2 = (const float*)d_in[15];
  const float* bb2 = (const float*)d_in[16];
  float* out = (float*)d_out;

  char* ws = (char*)d_ws;
  bf16_t* DB   = (bf16_t*)(ws + OFF_DB);
  bf16_t* SB   = (bf16_t*)(ws + OFF_SB);
  float*  PRE  = (float*)(ws + OFF_PRE);
  bf16_t* XCAT = (bf16_t*)(ws + OFF_XCAT);
  float*  HPRE = (float*)(ws + OFF_HPRE);
  bf16_t* HB   = (bf16_t*)(ws + OFF_HB);

  // 1) bf16 activation copies
  f32_to_bf16_kernel<<<(BATCH * DDIM / 4 + 255) / 256, 256, 0, stream>>>(deter, DB, BATCH * DDIM);
  f32_to_bf16_kernel<<<(BATCH * SDIM / 4 + 255) / 256, 256, 0, stream>>>(stoch, SB, BATCH * SDIM);

  // 2) action branch (K=2) -> PRE[:,2048:3072]
  action_branch_kernel<<<BATCH * HDIM / 256, 256, 0, stream>>>(action, W3, b3, PRE);

  // 3) x1 = d @ W1.T + b1 -> PRE[:,0:1024] ; x2 = s @ W2.T + b2 -> PRE[:,1024:2048]
  gemm_bf16_nt_kernel<<<dim3(HDIM / BN, BATCH / BM), 256, 0, stream>>>(
      DB, DDIM, W1, DDIM, b1, PRE + 0, 3072, DDIM);
  gemm_bf16_nt_kernel<<<dim3(HDIM / BN, BATCH / BM), 256, 0, stream>>>(
      SB, SDIM, W2, SDIM, b2, PRE + 1024, 3072, SDIM);

  // 4) per-branch RMSNorm + SiLU -> XCAT (bf16)
  rmsnorm_silu_kernel<<<BATCH, 256, 0, stream>>>(PRE + 0,    s1, XCAT + 0,    HDIM, 3072, 3072);
  rmsnorm_silu_kernel<<<BATCH, 256, 0, stream>>>(PRE + 1024, s2, XCAT + 1024, HDIM, 3072, 3072);
  rmsnorm_silu_kernel<<<BATCH, 256, 0, stream>>>(PRE + 2048, s3, XCAT + 2048, HDIM, 3072, 3072);

  // 5) grouped GEMM Wb1 -> HPRE (f32, +bb1)
  gemm_grouped_wb1_kernel<<<dim3(1024 / BN, BATCH / BM, GRP), 256, 0, stream>>>(
      XCAT, DB, Wb1, bb1, HPRE);

  // 6) RMSNorm(sb1) + SiLU over D=8192 -> HB (bf16)
  rmsnorm_silu_kernel<<<BATCH, 256, 0, stream>>>(HPRE, sb1, HB, DDIM, DDIM, DDIM);

  // 7) fused Wb2 GEMM + GRU gate -> out
  gemm_gate_wb2_kernel<<<dim3(1024 / BN, BATCH / BM, GRP), 256, 0, stream>>>(
      HB, Wb2, bb2, deter, out);
}
